// CausalSelfAttention_89275190215320
// MI455X (gfx1250) — matrix operand order repaired and checked by reading, not compiled, not run on hardware
//
#include <hip/hip_runtime.h>
#include <hip/hip_bf16.h>

// ---------------------------------------------------------------------------
// CDNA5 (gfx1250) causal self-attention: bf16 WMMA everywhere, f32 accum.
// Round 2: software-pipelined GEMMs (32x64/wave), LDS-tiled transpose,
// attention with 32 q-rows/wave + early V loads + global prefetch.
// ---------------------------------------------------------------------------

typedef __attribute__((ext_vector_type(16))) __bf16 v16bf;
typedef __attribute__((ext_vector_type(8)))  float  v8f;

#define S_LEN 4096
#define D_DIM 1024
#define NHEAD 16
#define HDIM  64

union FragU { uint4 u[2]; v16bf v; };

// A-fragment 16x32 bf16 (rows M=lane&15; lane<16 holds K 0..7,16..23; lane>=16 holds K 8..15,24..31)
__device__ inline v16bf frag_a(const __bf16* base, int ld, int lane) {
    const int half = lane >> 4, m = lane & 15;
    const __bf16* p = base + (size_t)m * ld + half * 8;
    FragU f;
    f.u[0] = *reinterpret_cast<const uint4*>(p);
    f.u[1] = *reinterpret_cast<const uint4*>(p + 16);
    return f.v;
}

// B-fragment 32x16 bf16: lane supplies column n=lane&15; K-dim contiguous in memory
__device__ inline v16bf frag_b(const __bf16* base, int ld, int lane) {
    const int half = lane >> 4, n = lane & 15;
    const __bf16* p = base + (size_t)n * ld + half * 8;
    FragU f;
    f.u[0] = *reinterpret_cast<const uint4*>(p);
    f.u[1] = *reinterpret_cast<const uint4*>(p + 16);
    return f.v;
}

__device__ inline v8f wmma_bf16(v16bf a, v16bf b, v8f c) {
    return __builtin_amdgcn_wmma_f32_16x16x32_bf16(false, a, false, b, (short)0, c, false, false);
}

// ---------------------------------------------------------------------------
// Converts
// ---------------------------------------------------------------------------
__global__ void k_f32_to_bf16_v4(const float4* __restrict__ in, __bf16* __restrict__ out, int n4) {
    int i = blockIdx.x * blockDim.x + threadIdx.x;
    if (i < n4) {
        float4 f = in[i];
        union { __bf16 h[4]; uint2 u; } p;
        p.h[0] = (__bf16)f.x; p.h[1] = (__bf16)f.y;
        p.h[2] = (__bf16)f.z; p.h[3] = (__bf16)f.w;
        *reinterpret_cast<uint2*>(out + 4 * (size_t)i) = p.u;
    }
}

// LDS-tiled transpose: in[R][C] f32 -> out[C][R] bf16, coalesced both ways.
// grid = (C/32, R/32), block = (32, 8)
__global__ __launch_bounds__(256) void k_transpose_to_bf16(
    const float* __restrict__ in, __bf16* __restrict__ out, int R, int C) {
    __shared__ float tile[32][33];
    const int bx = blockIdx.x * 32;   // col base (input)
    const int by = blockIdx.y * 32;   // row base (input)
    const int tx = threadIdx.x, ty = threadIdx.y;
#pragma unroll
    for (int i = 0; i < 32; i += 8)
        tile[ty + i][tx] = in[(size_t)(by + ty + i) * C + bx + tx];
    __syncthreads();
#pragma unroll
    for (int i = 0; i < 32; i += 8)
        out[(size_t)(bx + ty + i) * R + by + tx] = (__bf16)tile[tx][ty + i];
}

// ---------------------------------------------------------------------------
// Pipelined GEMM core: wave computes 32(M) x 64(N), K-step 32, double-buffered
// fragments so loads for step k+1 are in flight during step k's wmmas.
// ---------------------------------------------------------------------------
struct GemmAcc { v8f c[2][4]; };

template <typename EPI>
__device__ inline void gemm_wave_32x64(const __bf16* __restrict__ A, int lda,
                                       const __bf16* __restrict__ Bt, int ldb,
                                       int m0, int n0, int Kdim, int lane,
                                       EPI epilogue) {
    GemmAcc g;
#pragma unroll
    for (int mi = 0; mi < 2; ++mi)
#pragma unroll
        for (int t = 0; t < 4; ++t) g.c[mi][t] = (v8f){};

    v16bf a_cur[2], b_cur[4];
#pragma unroll
    for (int mi = 0; mi < 2; ++mi)
        a_cur[mi] = frag_a(A + (size_t)(m0 + mi * 16) * lda, lda, lane);
#pragma unroll
    for (int t = 0; t < 4; ++t)
        b_cur[t] = frag_b(Bt + (size_t)(n0 + t * 16) * ldb, ldb, lane);

    for (int k0 = 32; k0 < Kdim; k0 += 32) {
        v16bf a_nxt[2], b_nxt[4];
#pragma unroll
        for (int mi = 0; mi < 2; ++mi)
            a_nxt[mi] = frag_a(A + (size_t)(m0 + mi * 16) * lda + k0, lda, lane);
#pragma unroll
        for (int t = 0; t < 4; ++t)
            b_nxt[t] = frag_b(Bt + (size_t)(n0 + t * 16) * ldb + k0, ldb, lane);
#pragma unroll
        for (int mi = 0; mi < 2; ++mi)
#pragma unroll
            for (int t = 0; t < 4; ++t)
                g.c[mi][t] = wmma_bf16(a_cur[mi], b_cur[t], g.c[mi][t]);
#pragma unroll
        for (int mi = 0; mi < 2; ++mi) a_cur[mi] = a_nxt[mi];
#pragma unroll
        for (int t = 0; t < 4; ++t) b_cur[t] = b_nxt[t];
    }
#pragma unroll
    for (int mi = 0; mi < 2; ++mi)
#pragma unroll
        for (int t = 0; t < 4; ++t)
            g.c[mi][t] = wmma_bf16(a_cur[mi], b_cur[t], g.c[mi][t]);

    epilogue(g);
}

// ---------------------------------------------------------------------------
// QKV GEMM: scatter into Q[h][s][hd] (scaled), K[h][s][hd], Vt[h][hd][s].
// grid = (3072/64, 4096/256), block = 256 (8 waves, each 32M x 64N)
// ---------------------------------------------------------------------------
__global__ __launch_bounds__(256) void k_gemm_qkv(
    const __bf16* __restrict__ A, const __bf16* __restrict__ Bt,
    const float* __restrict__ bias,
    __bf16* __restrict__ Qo, __bf16* __restrict__ Ko, __bf16* __restrict__ Vto) {
    const int lane = threadIdx.x & 31;
    const int w    = threadIdx.x >> 5;
    const int half = lane >> 4, ln = lane & 15;
    const int m0 = blockIdx.y * 256 + w * 32;
    const int n0 = blockIdx.x * 64;

    gemm_wave_32x64(A, D_DIM, Bt, D_DIM, m0, n0, D_DIM, lane, [&](GemmAcc& g) {
#pragma unroll
        for (int t = 0; t < 4; ++t) {
            const int jj     = n0 + t * 16;
            const int region = jj >> 10;         // 0=Q 1=K 2=V (uniform per tile)
            const int jr     = jj & 1023;
            const int h      = jr >> 6;
            const int hd     = (jr & 63) + ln;
            const float bj   = bias[jj + ln];
#pragma unroll
            for (int mi = 0; mi < 2; ++mi)
#pragma unroll
                for (int r = 0; r < 8; ++r) {
                    const int s = m0 + mi * 16 + r + half * 8;
                    float val = g.c[mi][t][r] + bj;
                    if (region == 0) {
                        Qo[((size_t)h * S_LEN + s) * HDIM + hd] = (__bf16)(val * 0.125f);
                    } else if (region == 1) {
                        Ko[((size_t)h * S_LEN + s) * HDIM + hd] = (__bf16)val;
                    } else {
                        Vto[((size_t)h * HDIM + hd) * S_LEN + s] = (__bf16)val;
                    }
                }
        }
    });
}

// ---------------------------------------------------------------------------
// Output projection: out = O @ W_out + b_out (f32).
// grid = (1024/64, 4096/256), block = 256
// ---------------------------------------------------------------------------
__global__ __launch_bounds__(256) void k_gemm_out(
    const __bf16* __restrict__ A, const __bf16* __restrict__ Bt,
    const float* __restrict__ bias, float* __restrict__ out) {
    const int lane = threadIdx.x & 31;
    const int w    = threadIdx.x >> 5;
    const int half = lane >> 4, ln = lane & 15;
    const int m0 = blockIdx.y * 256 + w * 32;
    const int n0 = blockIdx.x * 64;

    gemm_wave_32x64(A, D_DIM, Bt, D_DIM, m0, n0, D_DIM, lane, [&](GemmAcc& g) {
#pragma unroll
        for (int t = 0; t < 4; ++t) {
            const int j = n0 + t * 16 + ln;
            const float bj = bias[j];
#pragma unroll
            for (int mi = 0; mi < 2; ++mi)
#pragma unroll
                for (int r = 0; r < 8; ++r) {
                    const int s = m0 + mi * 16 + r + half * 8;
                    out[(size_t)s * D_DIM + j] = g.c[mi][t][r] + bj;
                }
        }
    });
}

// ---------------------------------------------------------------------------
// Flash attention (causal): one wave owns 32 query rows of one head.
// K/V fragments reused across both 16-row q-subtiles (2x wmma density).
// V fragments issued before softmax so their latency hides under VALU.
// Block = 4 waves, each an independent q-tile (no block-wide sync).
// grid = (S/128, H), block = 128
// ---------------------------------------------------------------------------
__global__ __launch_bounds__(128) void k_flash_attn(
    const __bf16* __restrict__ Q, const __bf16* __restrict__ K,
    const __bf16* __restrict__ Vt, __bf16* __restrict__ O) {
    const int lane = threadIdx.x & 31;
    const int w    = threadIdx.x >> 5;
    const int half = lane >> 4, ln = lane & 15;
    const int h  = blockIdx.y;
    const int q0 = (blockIdx.x * 4 + w) * 32;

    __shared__ alignas(16) __bf16 Pbuf[4][2][16 * 32];

    // Q fragments: two 16-row subtiles x two K-steps over HD=64 (scale folded)
    v16bf qf[2][2];
#pragma unroll
    for (int mi = 0; mi < 2; ++mi) {
        const __bf16* qb = Q + ((size_t)h * S_LEN + q0 + mi * 16) * HDIM;
        qf[mi][0] = frag_a(qb, HDIM, lane);
        qf[mi][1] = frag_a(qb + 32, HDIM, lane);
    }

    v8f o[2][4];
#pragma unroll
    for (int mi = 0; mi < 2; ++mi)
#pragma unroll
        for (int t = 0; t < 4; ++t) o[mi][t] = (v8f){};
    float m_i[2][8], l_i[2][8], corr[2][8];
#pragma unroll
    for (int mi = 0; mi < 2; ++mi)
#pragma unroll
        for (int r = 0; r < 8; ++r) { m_i[mi][r] = -1e38f; l_i[mi][r] = 0.0f; }

    const int nkb = q0 / 32 + 1;           // keys <= q0+31
    for (int kb = 0; kb < nkb; ++kb) {
        const int kbase = kb * 32;
        const __bf16* kp = K + ((size_t)h * S_LEN + kbase) * HDIM;

        // prefetch next k-block's K and V lines (global_prefetch_b8)
        if (kb + 1 < nkb) {
            const __bf16* kn = kp + 32 * HDIM;
            __builtin_prefetch(kn + (size_t)lane * 128, 0, 3);
            const __bf16* vn = Vt + (size_t)h * HDIM * S_LEN + kbase + 32;
            __builtin_prefetch(vn + (size_t)(lane * 2) * S_LEN, 0, 3);
        }

        // ---- K fragments, then scores for both q-subtiles ----
        v16bf kf[2][2];
#pragma unroll
        for (int nt = 0; nt < 2; ++nt) {
            kf[nt][0] = frag_b(kp + (size_t)nt * 16 * HDIM, HDIM, lane);
            kf[nt][1] = frag_b(kp + (size_t)nt * 16 * HDIM + 32, HDIM, lane);
        }
        v8f s[2][2];
#pragma unroll
        for (int mi = 0; mi < 2; ++mi)
#pragma unroll
            for (int nt = 0; nt < 2; ++nt) {
                v8f c = (v8f){};
                c = wmma_bf16(qf[mi][0], kf[nt][0], c);
                c = wmma_bf16(qf[mi][1], kf[nt][1], c);
                s[mi][nt] = c;
            }

        // ---- V fragments early: latency hides under softmax VALU ----
        v16bf vf[4];
#pragma unroll
        for (int t = 0; t < 4; ++t)
            vf[t] = frag_b(Vt + ((size_t)h * HDIM + t * 16) * S_LEN + kbase, S_LEN, lane);

        // ---- causal mask on the diagonal block ----
        if (kb == nkb - 1) {
#pragma unroll
            for (int mi = 0; mi < 2; ++mi)
#pragma unroll
                for (int nt = 0; nt < 2; ++nt) {
                    const int colg = kbase + nt * 16 + ln;
#pragma unroll
                    for (int r = 0; r < 8; ++r) {
                        const int rowg = q0 + mi * 16 + r + half * 8;
                        if (colg > rowg) s[mi][nt][r] = -1e38f;
                    }
                }
        }

        // ---- online softmax (row reductions within 16-lane halves) ----
#pragma unroll
        for (int mi = 0; mi < 2; ++mi) {
            __bf16* pb = &Pbuf[w][mi][0];
#pragma unroll
            for (int r = 0; r < 8; ++r) {
                float v0 = s[mi][0][r], v1 = s[mi][1][r];
                float rm = fmaxf(v0, v1);
#pragma unroll
                for (int off = 1; off < 16; off <<= 1)
                    rm = fmaxf(rm, __shfl_xor(rm, off, 16));
                const float mn = fmaxf(m_i[mi][r], rm);
                const float e0 = __expf(v0 - mn);
                const float e1 = __expf(v1 - mn);
                float rs = e0 + e1;
#pragma unroll
                for (int off = 1; off < 16; off <<= 1)
                    rs += __shfl_xor(rs, off, 16);
                const float cr = __expf(m_i[mi][r] - mn);
                l_i[mi][r] = l_i[mi][r] * cr + rs;
                m_i[mi][r] = mn;
                corr[mi][r] = cr;
                const int row = r + half * 8;
                pb[row * 32 + ln]      = (__bf16)e0;
                pb[row * 32 + 16 + ln] = (__bf16)e1;
            }
        }

        // rescale running output
#pragma unroll
        for (int mi = 0; mi < 2; ++mi)
#pragma unroll
            for (int t = 0; t < 4; ++t)
#pragma unroll
                for (int r = 0; r < 8; ++r) o[mi][t][r] *= corr[mi][r];

        // wave-private LDS RAW: wait for DS stores before transposing reload
        asm volatile("s_wait_dscnt 0x0" ::: "memory");

#pragma unroll
        for (int mi = 0; mi < 2; ++mi) {
            v16bf pf = frag_a(&Pbuf[w][mi][0], 32, lane);
#pragma unroll
            for (int t = 0; t < 4; ++t)
                o[mi][t] = wmma_bf16(pf, vf[t], o[mi][t]);
        }
    }

    // ---- normalize & write O[s][h*64+hd] as bf16 ----
#pragma unroll
    for (int mi = 0; mi < 2; ++mi) {
        float invl[8];
#pragma unroll
        for (int r = 0; r < 8; ++r) invl[r] = 1.0f / l_i[mi][r];
#pragma unroll
        for (int t = 0; t < 4; ++t) {
            const int col = h * HDIM + t * 16 + ln;
#pragma unroll
            for (int r = 0; r < 8; ++r) {
                const int srow = q0 + mi * 16 + r + half * 8;
                O[(size_t)srow * D_DIM + col] = (__bf16)(o[mi][t][r] * invl[mi == 0 ? r : r]);
            }
        }
    }
}

// ---------------------------------------------------------------------------
// Launch
// ---------------------------------------------------------------------------
extern "C" void kernel_launch(void* const* d_in, const int* in_sizes, int n_in,
                              void* d_out, int out_size, void* d_ws, size_t ws_size,
                              hipStream_t stream) {
    const float* x     = (const float*)d_in[0];   // [1,4096,1024]
    const float* W_qkv = (const float*)d_in[1];   // [1024,3072]
    const float* b_qkv = (const float*)d_in[2];   // [3072]
    const float* W_out = (const float*)d_in[3];   // [1024,1024]
    const float* b_out = (const float*)d_in[4];   // [1024]
    float* out = (float*)d_out;                   // [1,4096,1024] f32

    char* ws = (char*)d_ws;
    const size_t MB = 1024u * 1024u;
    __bf16* x_bf   = (__bf16*)(ws + 0 * MB);    // 8 MB   4096x1024
    __bf16* wqkv_t = (__bf16*)(ws + 8 * MB);    // 6 MB   3072x1024 (W^T)
    __bf16* wout_t = (__bf16*)(ws + 14 * MB);   // 2 MB   1024x1024 (W^T)
    __bf16* Qb     = (__bf16*)(ws + 16 * MB);   // 8 MB   [16][4096][64], pre-scaled
    __bf16* Kb     = (__bf16*)(ws + 24 * MB);   // 8 MB   [16][4096][64]
    __bf16* Vtb    = (__bf16*)(ws + 32 * MB);   // 8 MB   [16][64][4096]
    __bf16* Ob     = (__bf16*)(ws + 40 * MB);   // 8 MB   [4096][1024]

    {   // converts
        int n4 = S_LEN * D_DIM / 4;
        k_f32_to_bf16_v4<<<n4 / 256, 256, 0, stream>>>((const float4*)x, x_bf, n4);
        k_transpose_to_bf16<<<dim3(3 * D_DIM / 32, D_DIM / 32), dim3(32, 8), 0, stream>>>(
            W_qkv, wqkv_t, D_DIM, 3 * D_DIM);
        k_transpose_to_bf16<<<dim3(D_DIM / 32, D_DIM / 32), dim3(32, 8), 0, stream>>>(
            W_out, wout_t, D_DIM, D_DIM);
    }

    // QKV projection: grid = (48, 16)
    k_gemm_qkv<<<dim3(3 * D_DIM / 64, S_LEN / 256), 256, 0, stream>>>(
        x_bf, wqkv_t, b_qkv, Qb, Kb, Vtb);

    // Flash attention: grid = (32, 16), 4 waves/block, 32 q-rows per wave
    k_flash_attn<<<dim3(S_LEN / 128, NHEAD), 128, 0, stream>>>(Qb, Kb, Vtb, Ob);

    // Output projection: grid = (16, 16)
    k_gemm_out<<<dim3(D_DIM / 64, S_LEN / 256), 256, 0, stream>>>(
        Ob, wout_t, b_out, out);
}
